// ReKT_concept_8589934592387
// MI455X (gfx1250) — matrix operand order repaired and checked by reading, not compiled, not run on hardware
//
#include <hip/hip_runtime.h>
#include <hip/hip_bf16.h>
#include <stdint.h>

#define BB 512
#define SS 199
#define DD 512
#define SKMAX 300
#define TSLOTS 200
#define PITCH (DD + 8)

typedef _Float16 f16;
typedef _Float16 v16h __attribute__((ext_vector_type(16)));
typedef float v8f __attribute__((ext_vector_type(8)));

union Frag {
  v16h v;
  uint4 q[2];
};

__device__ __forceinline__ v8f wmma_f16(v16h a, v16h b, v8f c) {
  // D = A(16x32 f16) * B(32x16 f16) + C(16x16 f32)
  return __builtin_amdgcn_wmma_f32_16x16x32_f16(false, a, false, b, (short)0, c,
                                                false, false);
}

// Fast transcendentals via v_exp_f32 + v_rcp_f32 (no div_scale sequences).
__device__ __forceinline__ float fast_rcp(float x) {
  return __builtin_amdgcn_rcpf(x);
}
__device__ __forceinline__ float sigmoidf_(float x) {
  // exp(-x) -> inf  => rcp -> 0 ; exp(-x) -> 0 => 1.  Saturates safely.
  return fast_rcp(1.f + __expf(-x));
}
__device__ __forceinline__ float tanhf_(float x) {
  // 1 - 2/(e^{2x}+1): overflow-safe at both ends (inf -> 1, 0 -> -1).
  return 1.f - 2.f * fast_rcp(__expf(2.f * x) + 1.f);
}

// ---------------------------------------------------------------------------
// hist[b][s][c] = f16(skill_state0[s][c])   (replicates skill_buf0 init)
// ---------------------------------------------------------------------------
__global__ void k_init_hist(const float* __restrict__ skill_state0,
                            f16* __restrict__ hist) {
  size_t idx = (size_t)blockIdx.x * blockDim.x + threadIdx.x; // one 8-f16 chunk
  size_t total = (size_t)BB * SS * DD / 8;
  if (idx >= total) return;
  size_t f = idx * 8;
  int c = (int)(f % DD);
  int s = (int)((f / DD) % SS);
  const float* src = skill_state0 + (size_t)s * DD + c;
  f16 tmp[8];
#pragma unroll
  for (int i = 0; i < 8; ++i) tmp[i] = (f16)src[i];
  *(uint4*)(hist + f) = *(const uint4*)tmp;
}

// ---------------------------------------------------------------------------
// Pack 6 [512x512] f32 weight blocks into per-lane WMMA B-fragment order:
// Wpk[m][ct][ks][lane][16 f16]  ->  j<8: K=slab+8*(lane>>4)+j, j>=8: K=slab+16+8*(lane>>4)+(j-8)
// m: 0=W_sf top, 1=W_af top, 2=W_ss top, 3=W_as top, 4=W_o1 rows[0:512),
//    5=W_o1 rows[512:1024)
// ---------------------------------------------------------------------------
__global__ void k_pack(const float* __restrict__ Wsf, const float* __restrict__ Waf,
                       const float* __restrict__ Wss, const float* __restrict__ Was,
                       const float* __restrict__ Wo1, f16* __restrict__ Wpk) {
  int idx = blockIdx.x * blockDim.x + threadIdx.x;
  if (idx >= 6 * DD * DD) return;
  int m = idx / (DD * DD);
  int rem = idx % (DD * DD);
  int j2 = rem & 15;
  int lane = (rem >> 4) & 31;
  int ks = (rem >> 9) & 15;
  int ct = (rem >> 13) & 31;
  int n = ct * 16 + (lane & 15);
  int k = ks * 32 + ((lane >> 4) << 3) + (j2 & 7) + ((j2 >> 3) << 4);
  const float* src;
  int roff = 0;
  switch (m) {
    case 0: src = Wsf; break;
    case 1: src = Waf; break;
    case 2: src = Wss; break;
    case 3: src = Was; break;
    case 4: src = Wo1; break;
    default: src = Wo1; roff = DD; break;
  }
  Wpk[idx] = (f16)src[(size_t)(roff + k) * DD + n];
}

// ---------------------------------------------------------------------------
// out[r][c] = sum_k inA[r][k] * W[woff+k][c] (+ bias[c])   (tiny, once/launch)
// ---------------------------------------------------------------------------
__global__ void k_proj(const float* __restrict__ inA, const float* __restrict__ W,
                       int woff, const float* __restrict__ bias, int R,
                       float* __restrict__ out) {
  int idx = blockIdx.x * blockDim.x + threadIdx.x;
  if (idx >= R * DD) return;
  int r = idx / DD, c = idx % DD;
  const float* a = inA + (size_t)r * DD;
  const float* w = W + (size_t)woff * DD + c;
  float acc = bias ? bias[c] : 0.f;
  for (int k = 0; k < DD; ++k) acc = fmaf(a[k], w[(size_t)k * DD], acc);
  out[idx] = acc;
}

// ---------------------------------------------------------------------------
// Main recurrent kernel: 32 blocks x 16 batch rows, 8 waves x 4 col-tiles.
// ---------------------------------------------------------------------------
__launch_bounds__(256, 1)
__global__ void k_rekt(const int* __restrict__ next_skill,
                       const int* __restrict__ next_ans,
                       const float* __restrict__ ls_state,
                       const float* __restrict__ Tsf,     // [200][512] gap proj + b_sf
                       const float* __restrict__ c_af,    // [512] time1 proj + b_af
                       const float* __restrict__ E_ss,    // [300][512]
                       const float* __restrict__ A2_ss,   // [2][512] (+b_ss)
                       const float* __restrict__ E_as,    // [300][512]
                       const float* __restrict__ A2_as,   // [2][512] (+b_as)
                       const float* __restrict__ Eo1c,    // [300][512] (+b_o1)
                       const float* __restrict__ Wo2,     // [512]
                       const float* __restrict__ b_o2,    // [1]
                       const f16* __restrict__ Wpk,       // packed weights
                       f16* __restrict__ hist,            // [B][S][512] f16
                       float* __restrict__ out) {         // [B][S]
  __shared__ f16 tSK[16][PITCH];
  __shared__ f16 tALL[16][PITCH];
  __shared__ int s_sk[16], s_tl[16], s_gap[16], s_ans[16];
  __shared__ float s_P[16];
  __shared__ int s_lt[16 * SKMAX];

  const int tid = threadIdx.x;
  const int lane = tid & 31;
  const int w = tid >> 5;
  const int hl = lane >> 4;
  const int ln = lane & 15;
  const int arow = ln;           // A-fragment row for this lane
  const int rb = blockIdx.x * 16;
  const uint4* Wq = (const uint4*)Wpk;
  const v8f vzero = {0.f, 0.f, 0.f, 0.f, 0.f, 0.f, 0.f, 0.f};

  for (int k = tid; k < 16 * SKMAX; k += 256) s_lt[k] = 0;

  // all_state registers, accumulator layout: allA[i][j] = state[row=j+8*hl][col]
  v8f allA[4];
#pragma unroll
  for (int i = 0; i < 4; ++i) {
    float v = ls_state[w * 64 + i * 16 + ln]; // broadcast over batch rows
#pragma unroll
    for (int j = 0; j < 8; ++j) allA[i][j] = v;
  }
  __syncthreads();

  for (int t = 0; t < SS; ++t) {
    // ---- stage 1: per-row scalars -------------------------------------
    if (tid < 16) {
      int b = rb + tid;
      int sk = next_skill[(size_t)b * SS + t];
      int tl = s_lt[tid * SKMAX + sk];
      s_sk[tid] = sk;
      s_tl[tid] = tl;
      s_gap[tid] = t - tl;
      s_ans[tid] = next_ans[(size_t)b * SS + t];
      s_P[tid] = 0.f;
    }
    __syncthreads();

    // ---- stage 2: gather last_sk rows -> tSK, dump all_state -> tALL --
#pragma unroll
    for (int c = 0; c < 4; ++c) {
      int idx = tid + c * 256;
      int row = idx >> 6, seg = idx & 63;
      const uint4* src =
          (const uint4*)(hist + ((size_t)(rb + row) * SS + s_tl[row]) * DD) + seg;
      *(uint4*)&tSK[row][seg * 8] = *src;
    }
#pragma unroll
    for (int i = 0; i < 4; ++i) {
      int col = w * 64 + i * 16 + ln;
#pragma unroll
      for (int j = 0; j < 8; ++j) tALL[j + 8 * hl][col] = (f16)allA[i][j];
    }
    __syncthreads();

    // ---- stage 3: forget-gate GEMMs: S1 = sk@Wsf_top, S2 = all@Waf_top
    v8f aS1[4], aS2[4];
#pragma unroll
    for (int i = 0; i < 4; ++i) { aS1[i] = vzero; aS2[i] = vzero; }
    for (int ks = 0; ks < 16; ++ks) {
      int c0 = ks * 32 + (hl << 3);
      Frag fsk, fal;
      fsk.q[0] = *(const uint4*)&tSK[arow][c0];
      fsk.q[1] = *(const uint4*)&tSK[arow][c0 + 16];
      fal.q[0] = *(const uint4*)&tALL[arow][c0];
      fal.q[1] = *(const uint4*)&tALL[arow][c0 + 16];
#pragma unroll
      for (int i = 0; i < 4; ++i) {
        int ct = w * 4 + i;
        Frag b0, b1;
        int i0 = (((0 * 32 + ct) * 16 + ks) * 32 + lane) * 2;
        int i1 = (((1 * 32 + ct) * 16 + ks) * 32 + lane) * 2;
        b0.q[0] = Wq[i0]; b0.q[1] = Wq[i0 + 1];
        b1.q[0] = Wq[i1]; b1.q[1] = Wq[i1 + 1];
        aS1[i] = wmma_f16(fsk.v, b0.v, aS1[i]);
        aS2[i] = wmma_f16(fal.v, b1.v, aS2[i]);
      }
    }
    __syncthreads(); // all GEMM reads of tiles done before overwrite

    // ---- stage 4: apply gates, rewrite tiles with gated f16 states ----
#pragma unroll
    for (int i = 0; i < 4; ++i) {
      int col = w * 64 + i * 16 + ln;
#pragma unroll
      for (int j = 0; j < 8; ++j) {
        int row = j + 8 * hl;
        float raw = (float)tSK[row][col];
        float fsk = sigmoidf_(aS1[i][j] + Tsf[(size_t)s_gap[row] * DD + col]);
        float lsk = raw * fsk;
        float fall = sigmoidf_(aS2[i][j] + c_af[col]);
        float lall = allA[i][j] * fall;
        aS1[i][j] = lsk;   // keep f32 last_sk
        aS2[i][j] = lall;  // keep f32 last_all
        tSK[row][col] = (f16)lsk;
        tALL[row][col] = (f16)lall;
      }
    }
    __syncthreads();

    // ---- stage 5a: output head: O = lall@Wo1a + lsk@Wo1b --------------
    v8f aO[4];
#pragma unroll
    for (int i = 0; i < 4; ++i) aO[i] = vzero;
    for (int ks = 0; ks < 16; ++ks) {
      int c0 = ks * 32 + (hl << 3);
      Frag fsk, fal;
      fal.q[0] = *(const uint4*)&tALL[arow][c0];
      fal.q[1] = *(const uint4*)&tALL[arow][c0 + 16];
      fsk.q[0] = *(const uint4*)&tSK[arow][c0];
      fsk.q[1] = *(const uint4*)&tSK[arow][c0 + 16];
#pragma unroll
      for (int i = 0; i < 4; ++i) {
        int ct = w * 4 + i;
        Frag ba, bb;
        int ia = (((4 * 32 + ct) * 16 + ks) * 32 + lane) * 2;
        int ib = (((5 * 32 + ct) * 16 + ks) * 32 + lane) * 2;
        ba.q[0] = Wq[ia]; ba.q[1] = Wq[ia + 1];
        bb.q[0] = Wq[ib]; bb.q[1] = Wq[ib + 1];
        aO[i] = wmma_f16(fal.v, ba.v, aO[i]);
        aO[i] = wmma_f16(fsk.v, bb.v, aO[i]);
      }
    }
    // P partial: relu(O + Eo1c[sk]) . Wo2 reduced per row via LDS atomics
#pragma unroll
    for (int j = 0; j < 8; ++j) {
      int row = j + 8 * hl;
      int skr = s_sk[row];
      float part = 0.f;
#pragma unroll
      for (int i = 0; i < 4; ++i) {
        int col = w * 64 + i * 16 + ln;
        float h = aO[i][j] + Eo1c[(size_t)skr * DD + col];
        part += fmaxf(h, 0.f) * Wo2[col];
      }
      atomicAdd(&s_P[row], part);
    }

    // ---- stage 5b: new_all = lall + tanh(lall@Was_top + Xas) ----------
    v8f aT[4];
#pragma unroll
    for (int i = 0; i < 4; ++i) aT[i] = vzero;
    for (int ks = 0; ks < 16; ++ks) {
      int c0 = ks * 32 + (hl << 3);
      Frag fal;
      fal.q[0] = *(const uint4*)&tALL[arow][c0];
      fal.q[1] = *(const uint4*)&tALL[arow][c0 + 16];
#pragma unroll
      for (int i = 0; i < 4; ++i) {
        int ct = w * 4 + i;
        Frag b;
        int ib = (((3 * 32 + ct) * 16 + ks) * 32 + lane) * 2;
        b.q[0] = Wq[ib]; b.q[1] = Wq[ib + 1];
        aT[i] = wmma_f16(fal.v, b.v, aT[i]);
      }
    }
#pragma unroll
    for (int i = 0; i < 4; ++i) {
      int col = w * 64 + i * 16 + ln;
#pragma unroll
      for (int j = 0; j < 8; ++j) {
        int row = j + 8 * hl;
        float x = aT[i][j] + E_as[(size_t)s_sk[row] * DD + col] +
                  A2_as[(size_t)s_ans[row] * DD + col];
        allA[i][j] = aS2[i][j] + tanhf_(x);
      }
    }

    // ---- stage 5c: new_sk = lsk + tanh(lsk@Wss_top + Xss) -> hist[t] --
#pragma unroll
    for (int i = 0; i < 4; ++i) aT[i] = vzero;
    for (int ks = 0; ks < 16; ++ks) {
      int c0 = ks * 32 + (hl << 3);
      Frag fsk;
      fsk.q[0] = *(const uint4*)&tSK[arow][c0];
      fsk.q[1] = *(const uint4*)&tSK[arow][c0 + 16];
#pragma unroll
      for (int i = 0; i < 4; ++i) {
        int ct = w * 4 + i;
        Frag b;
        int ib = (((2 * 32 + ct) * 16 + ks) * 32 + lane) * 2;
        b.q[0] = Wq[ib]; b.q[1] = Wq[ib + 1];
        aT[i] = wmma_f16(fsk.v, b.v, aT[i]);
      }
    }
#pragma unroll
    for (int i = 0; i < 4; ++i) {
      int col = w * 64 + i * 16 + ln;
#pragma unroll
      for (int j = 0; j < 8; ++j) {
        int row = j + 8 * hl;
        float x = aT[i][j] + E_ss[(size_t)s_sk[row] * DD + col] +
                  A2_ss[(size_t)s_ans[row] * DD + col];
        float ns = aS1[i][j] + tanhf_(x);
        hist[((size_t)(rb + row) * SS + t) * DD + col] = (f16)ns;
      }
    }
    __threadfence_block(); // hist RAW (next step may gather slot t)
    __syncthreads();

    // ---- stage 6: emit P, update last_time ----------------------------
    if (tid < 16) {
      out[(size_t)(rb + tid) * SS + t] = sigmoidf_(s_P[tid] + b_o2[0]);
      s_lt[tid * SKMAX + s_sk[tid]] = t;
    }
    __syncthreads();
  }
}

// ---------------------------------------------------------------------------
extern "C" void kernel_launch(void* const* d_in, const int* in_sizes, int n_in,
                              void* d_out, int out_size, void* d_ws, size_t ws_size,
                              hipStream_t stream) {
  (void)in_sizes; (void)n_in; (void)out_size; (void)ws_size;
  const int* next_skill = (const int*)d_in[4];
  const int* next_ans = (const int*)d_in[5];
  const float* skill_embed = (const float*)d_in[6];
  const float* ans_embed = (const float*)d_in[7];
  const float* time_embed = (const float*)d_in[8];
  const float* ls_state = (const float*)d_in[9];
  const float* skill_state0 = (const float*)d_in[10];
  const float* W_sf = (const float*)d_in[11];
  const float* b_sf = (const float*)d_in[12];
  const float* W_af = (const float*)d_in[13];
  const float* b_af = (const float*)d_in[14];
  const float* W_ss = (const float*)d_in[15];
  const float* b_ss = (const float*)d_in[16];
  const float* W_as = (const float*)d_in[17];
  const float* b_as = (const float*)d_in[18];
  const float* W_o1 = (const float*)d_in[19];
  const float* b_o1 = (const float*)d_in[20];
  const float* W_o2 = (const float*)d_in[21];
  const float* b_o2 = (const float*)d_in[22];

  char* ws = (char*)d_ws;
  size_t off = 0;
  auto alloc = [&](size_t bytes) -> void* {
    void* p = ws + off;
    off = (off + bytes + 255) & ~(size_t)255;
    return p;
  };
  f16* hist = (f16*)alloc((size_t)BB * SS * DD * sizeof(f16));
  f16* Wpk = (f16*)alloc((size_t)6 * DD * DD * sizeof(f16));
  float* Tsf = (float*)alloc((size_t)TSLOTS * DD * sizeof(float));
  float* cafT = (float*)alloc((size_t)DD * sizeof(float));
  float* Ess = (float*)alloc((size_t)SKMAX * DD * sizeof(float));
  float* A2ss = (float*)alloc((size_t)2 * DD * sizeof(float));
  float* Eas = (float*)alloc((size_t)SKMAX * DD * sizeof(float));
  float* A2as = (float*)alloc((size_t)2 * DD * sizeof(float));
  float* Eo1c = (float*)alloc((size_t)SKMAX * DD * sizeof(float));

  size_t nchunks = (size_t)BB * SS * DD / 8;
  k_init_hist<<<(unsigned)((nchunks + 255) / 256), 256, 0, stream>>>(skill_state0, hist);
  k_pack<<<(6 * DD * DD + 255) / 256, 256, 0, stream>>>(W_sf, W_af, W_ss, W_as, W_o1, Wpk);
  k_proj<<<(TSLOTS * DD + 255) / 256, 256, 0, stream>>>(time_embed, W_sf, DD, b_sf, TSLOTS, Tsf);
  k_proj<<<(DD + 255) / 256, 256, 0, stream>>>(time_embed + DD, W_af, DD, b_af, 1, cafT);
  k_proj<<<(SKMAX * DD + 255) / 256, 256, 0, stream>>>(skill_embed, W_ss, DD, nullptr, SKMAX, Ess);
  k_proj<<<(2 * DD + 255) / 256, 256, 0, stream>>>(ans_embed, W_ss, DD, b_ss, 2, A2ss);
  k_proj<<<(SKMAX * DD + 255) / 256, 256, 0, stream>>>(skill_embed, W_as, DD, nullptr, SKMAX, Eas);
  k_proj<<<(2 * DD + 255) / 256, 256, 0, stream>>>(ans_embed, W_as, DD, b_as, 2, A2as);
  k_proj<<<(SKMAX * DD + 255) / 256, 256, 0, stream>>>(skill_embed, W_o1, 2 * DD, b_o1, SKMAX, Eo1c);

  k_rekt<<<BB / 16, 256, 0, stream>>>(next_skill, next_ans, ls_state, Tsf, cafT,
                                      Ess, A2ss, Eas, A2as, Eo1c, W_o2, b_o2,
                                      Wpk, hist, (float*)d_out);
}
